// MambaMIL_2D_71451075937042
// MI455X (gfx1250) — compile-verified, matrix operand order checked by reading, not run on hardware
//
#include <hip/hip_runtime.h>
#include <hip/hip_bf16.h>

#define NTOK   2048
#define INDIM  1024
#define DMODEL 512
#define DINNER 1024
#define DSTATE 16
#define DTRANK 32
#define DCONV  4
#define NLAYER 2
#define ATTND  128
#define CHUNK  32
#define NCH    (NTOK / CHUNK)   /* 64 chunks for the parallel scan */

typedef __attribute__((ext_vector_type(16))) _Float16 v16h;
typedef __attribute__((ext_vector_type(8)))  _Float16 v8h;
typedef __attribute__((ext_vector_type(8)))  float    v8f;
typedef __attribute__((ext_vector_type(4)))  float    v4f;

// ---------------------------------------------------------------------------
// device helpers
// ---------------------------------------------------------------------------
__device__ __forceinline__ float dev_silu(float x) { return x / (1.f + expf(-x)); }

// CDNA5 16-bit A/B fragment for 16x16x32: two contiguous 16B chunks per lane.
__device__ __forceinline__ v16h load_frag(const _Float16* __restrict__ p) {
    v8h lo = *(const v8h*)(p);
    v8h hi = *(const v8h*)(p + 16);
    v16h r;
#pragma unroll
    for (int i = 0; i < 8; ++i) { r[i] = lo[i]; r[i + 8] = hi[i]; }
    return r;
}

__device__ __forceinline__ void store_tile(float* __restrict__ C, const float* __restrict__ bias,
                                           v8f acc, int m0, int n0, int N, int lane, int accum) {
    const int cn  = n0 + (lane & 15);
    const int cmb = m0 + ((lane >> 4) << 3);
    const float bv = bias ? bias[cn] : 0.f;
#pragma unroll
    for (int r = 0; r < 8; ++r) {
        size_t idx = (size_t)(cmb + r) * N + cn;
        float v = acc[r] + bv;
        if (accum) C[idx] += v; else C[idx] = v;
    }
}

// Cooperative async stage of one chunk's B/C slice of dbc into LDS.
// 32 rows (timesteps) x 32 floats (B[16] then C[16]) = 4KB; 256 threads do
// one global_load_async_to_lds_b128 each, tracked by ASYNCcnt.
__device__ __forceinline__ void stage_bc_tile(const float* __restrict__ dbc, int c,
                                              float* __restrict__ sbc, int tid) {
    const int r  = tid >> 3;                 // 0..31 : timestep within chunk
    const int s4 = (tid & 7) << 2;           // 0,4,...,28 : dword quad
    const float* gsrc = dbc + ((size_t)(c * CHUNK + r) * 64 + DTRANK) + s4;
    float* ldst = sbc + r * 32 + s4;
    unsigned lds_off = (unsigned)(uintptr_t)ldst;
    asm volatile("global_load_async_to_lds_b128 %0, %1, off"
                 :: "v"(lds_off), "v"(gsrc)
                 : "memory");
    asm volatile("s_wait_asynccnt 0x0" ::: "memory");
    __syncthreads();
}

// ---------------------------------------------------------------------------
// Generic f16 WMMA GEMM: C[M,N](f32) (+)= A[M,K](f16,row) * BT[N,K](f16) + bias
// block = 128 threads = 4 waves; each wave owns a 32x32 tile (2x2 WMMA tiles),
// reusing both A fragments and both B fragments -> 4 wmma per k-step.
// Software-pipelined: next k-step fragments are loaded before current wmmas.
// Requires M % 32 == 0 and N % 32 == 0 (true for every GEMM in this model).
// ---------------------------------------------------------------------------
__global__ __launch_bounds__(128)
void k_gemm_f16(const _Float16* __restrict__ A, const _Float16* __restrict__ BT,
                float* __restrict__ C, const float* __restrict__ bias,
                int M, int N, int K, int accum)
{
    const int lane = threadIdx.x & 31;
    const int wave = threadIdx.x >> 5;
    const int m0   = blockIdx.y << 5;               // 32 rows per block-row
    const int row  = lane & 15;
    const int kb   = (lane >> 4) << 3;

    const int n0 = (blockIdx.x * 8 + wave * 2) << 4;  // wave covers n0..n0+31
    const int n1 = n0 + 16;
    if (n0 >= N) return;                              // wave-uniform

    const _Float16* ap0 = A  + (size_t)(m0 + row) * K + kb;
    const _Float16* ap1 = ap0 + (size_t)16 * K;
    const _Float16* bp0 = BT + (size_t)(n0 + row) * K + kb;
    const _Float16* bp1 = bp0 + (size_t)16 * K;

    v8f acc00 = {}; v8f acc01 = {}; v8f acc10 = {}; v8f acc11 = {};

    v16h a0 = load_frag(ap0), a1 = load_frag(ap1);
    v16h b0 = load_frag(bp0), b1 = load_frag(bp1);

    for (int k0 = 32; k0 < K; k0 += 32) {
        v16h a0n = load_frag(ap0 + k0);
        v16h a1n = load_frag(ap1 + k0);
        v16h b0n = load_frag(bp0 + k0);
        v16h b1n = load_frag(bp1 + k0);
        __builtin_prefetch(ap0 + k0 + 32, 0, 1);      // global_prefetch_b8
        __builtin_prefetch(bp0 + k0 + 32, 0, 1);
        acc00 = __builtin_amdgcn_wmma_f32_16x16x32_f16(false, a0, false, b0, (short)0, acc00, false, false);
        acc01 = __builtin_amdgcn_wmma_f32_16x16x32_f16(false, a0, false, b1, (short)0, acc01, false, false);
        acc10 = __builtin_amdgcn_wmma_f32_16x16x32_f16(false, a1, false, b0, (short)0, acc10, false, false);
        acc11 = __builtin_amdgcn_wmma_f32_16x16x32_f16(false, a1, false, b1, (short)0, acc11, false, false);
        a0 = a0n; a1 = a1n; b0 = b0n; b1 = b1n;
    }
    acc00 = __builtin_amdgcn_wmma_f32_16x16x32_f16(false, a0, false, b0, (short)0, acc00, false, false);
    acc01 = __builtin_amdgcn_wmma_f32_16x16x32_f16(false, a0, false, b1, (short)0, acc01, false, false);
    acc10 = __builtin_amdgcn_wmma_f32_16x16x32_f16(false, a1, false, b0, (short)0, acc10, false, false);
    acc11 = __builtin_amdgcn_wmma_f32_16x16x32_f16(false, a1, false, b1, (short)0, acc11, false, false);

    store_tile(C, bias, acc00, m0,      n0, N, lane, accum);
    store_tile(C, bias, acc01, m0,      n1, N, lane, accum);
    store_tile(C, bias, acc10, m0 + 16, n0, N, lane, accum);
    store_tile(C, bias, acc11, m0 + 16, n1, N, lane, accum);
}

// ---------------------------------------------------------------------------
// conversions
// ---------------------------------------------------------------------------
__global__ void k_cvt(const float* __restrict__ in, _Float16* __restrict__ out, int n) {
    int i = blockIdx.x * 256 + threadIdx.x;
    if (i < n) out[i] = (_Float16)in[i];
}

// weight f32[K,N] -> f16[N,K] (transpose so B fragments are contiguous)
__global__ void k_cvt_transpose(const float* __restrict__ in, _Float16* __restrict__ out,
                                int K, int N) {
    int i = blockIdx.x * 256 + threadIdx.x;
    if (i >= K * N) return;
    int k = i % K, n = i / K;
    out[(size_t)n * K + k] = (_Float16)in[(size_t)k * N + n];
}

// ---------------------------------------------------------------------------
// elementwise / normalization
// ---------------------------------------------------------------------------
__global__ void k_gelu_posadd(float* __restrict__ h, const float* __restrict__ pos,
                              const float* __restrict__ pos_w, const float* __restrict__ pos_b) {
    int i = blockIdx.x * 256 + threadIdx.x;        // NTOK*DMODEL
    int d = i & (DMODEL - 1), t = i >> 9;
    float v = h[i];
    v = 0.5f * v * (1.f + erff(v * 0.70710678f));  // exact GELU
    v += pos[t * 2] * pos_w[d] + pos[t * 2 + 1] * pos_w[DMODEL + d] + pos_b[d];
    h[i] = v;
}

__global__ __launch_bounds__(256)
void k_rmsnorm_f16(const float* __restrict__ h, const float* __restrict__ w,
                   _Float16* __restrict__ out) {
    __shared__ float red[256];
    const int t = blockIdx.x, tid = threadIdx.x;
    const float* row = h + (size_t)t * DMODEL;
    float s = 0.f;
    for (int d = tid; d < DMODEL; d += 256) { float v = row[d]; s += v * v; }
    red[tid] = s; __syncthreads();
    for (int off = 128; off > 0; off >>= 1) { if (tid < off) red[tid] += red[tid + off]; __syncthreads(); }
    float inv = rsqrtf(red[0] / (float)DMODEL + 1e-5f);
    for (int d = tid; d < DMODEL; d += 256)
        out[(size_t)t * DMODEL + d] = (_Float16)(row[d] * w[d] * inv);
}

__global__ __launch_bounds__(256)
void k_layernorm(const float* __restrict__ h, const float* __restrict__ w,
                 const float* __restrict__ b, float* __restrict__ outf,
                 _Float16* __restrict__ outh) {
    __shared__ float red[256];
    const int t = blockIdx.x, tid = threadIdx.x;
    const float* row = h + (size_t)t * DMODEL;
    float s = 0.f;
    for (int d = tid; d < DMODEL; d += 256) s += row[d];
    red[tid] = s; __syncthreads();
    for (int off = 128; off > 0; off >>= 1) { if (tid < off) red[tid] += red[tid + off]; __syncthreads(); }
    float mu = red[0] / (float)DMODEL; __syncthreads();
    float v = 0.f;
    for (int d = tid; d < DMODEL; d += 256) { float q = row[d] - mu; v += q * q; }
    red[tid] = v; __syncthreads();
    for (int off = 128; off > 0; off >>= 1) { if (tid < off) red[tid] += red[tid + off]; __syncthreads(); }
    float inv = rsqrtf(red[0] / (float)DMODEL + 1e-5f);
    for (int d = tid; d < DMODEL; d += 256) {
        float o = (row[d] - mu) * inv * w[d] + b[d];
        outf[(size_t)t * DMODEL + d] = o;
        outh[(size_t)t * DMODEL + d] = (_Float16)o;
    }
}

// causal depthwise conv1d (k=4) + SiLU, reading xin = xz[:, 0:DINNER]
__global__ void k_conv_silu(const float* __restrict__ xz, const float* __restrict__ cw,
                            const float* __restrict__ cb, float* __restrict__ u,
                            _Float16* __restrict__ uh) {
    int i = blockIdx.x * 256 + threadIdx.x;        // NTOK*DINNER
    int e = i & (DINNER - 1), t = i >> 10;
    float acc = cb[e];
#pragma unroll
    for (int k = 0; k < DCONV; ++k) {
        int tt = t - (DCONV - 1) + k;
        if (tt >= 0) acc += cw[e * DCONV + k] * xz[(size_t)tt * (2 * DINNER) + e];
    }
    float s = dev_silu(acc);
    u[i] = s; uh[i] = (_Float16)s;
}

__global__ void k_extract_dtr(const float* __restrict__ dbc, _Float16* __restrict__ dtrh) {
    int i = blockIdx.x * 256 + threadIdx.x;        // NTOK*DTRANK
    int r = i & (DTRANK - 1), t = i >> 5;
    dtrh[i] = (_Float16)dbc[(size_t)t * 64 + r];
}

__global__ void k_softplus(float* __restrict__ d, int n) {
    int i = blockIdx.x * 256 + threadIdx.x;
    if (i < n) { float v = d[i]; d[i] = (v > 20.f) ? v : log1pf(expf(v)); }
}

__global__ void k_tanh_ew(float* __restrict__ d, int n) {
    int i = blockIdx.x * 256 + threadIdx.x;
    if (i < n) d[i] = tanhf(d[i]);
}

// ---------------------------------------------------------------------------
// 3-phase chunked selective scan (diagonal affine composition).
// Each 256-thread block handles 256 channels of ONE chunk, so the chunk's
// B/C slice is staged once into LDS via async copy.
// ---------------------------------------------------------------------------
__global__ __launch_bounds__(256)
void k_scan_chunks(const float* __restrict__ delta, const float* __restrict__ u,
                   const float* __restrict__ dbc, const float* __restrict__ A_log,
                   float* __restrict__ chA, float* __restrict__ chB) {
    __shared__ __align__(16) float sbc[CHUNK * 32];
    int i = blockIdx.x * 256 + threadIdx.x;        // NCH*DINNER
    int e = i & (DINNER - 1), c = i >> 10;
    stage_bc_tile(dbc, c, sbc, threadIdx.x);

    float An[DSTATE], a[DSTATE], b[DSTATE];
#pragma unroll
    for (int n = 0; n < DSTATE; ++n) {
        An[n] = -expf(A_log[(size_t)e * DSTATE + n]);
        a[n] = 1.f; b[n] = 0.f;
    }
    const int tbase = c * CHUNK;
    for (int tt = 0; tt < CHUNK; ++tt) {
        const int t = tbase + tt;
        const float d  = delta[(size_t)t * DINNER + e];
        const float du = d * u[(size_t)t * DINNER + e];
        const float* Bv = sbc + tt * 32;           // B in LDS
#pragma unroll
        for (int n = 0; n < DSTATE; ++n) {
            float dA = expf(d * An[n]);
            a[n] *= dA;
            b[n] = dA * b[n] + du * Bv[n];
        }
    }
    size_t o = ((size_t)c * DINNER + e) * DSTATE;
#pragma unroll
    for (int n = 0; n < DSTATE; ++n) { chA[o + n] = a[n]; chB[o + n] = b[n]; }
}

__global__ void k_scan_prefix(const float* __restrict__ chA, const float* __restrict__ chB,
                              float* __restrict__ hin) {
    int e = blockIdx.x * 256 + threadIdx.x;
    if (e >= DINNER) return;
    float h[DSTATE];
#pragma unroll
    for (int n = 0; n < DSTATE; ++n) h[n] = 0.f;
    for (int c = 0; c < NCH; ++c) {
        size_t o = ((size_t)c * DINNER + e) * DSTATE;
#pragma unroll
        for (int n = 0; n < DSTATE; ++n) {
            hin[o + n] = h[n];                      // exclusive prefix (state entering chunk c)
            h[n] = chA[o + n] * h[n] + chB[o + n];
        }
    }
}

__global__ __launch_bounds__(256)
void k_scan_apply(const float* __restrict__ delta, const float* __restrict__ u,
                  const float* __restrict__ dbc, const float* __restrict__ A_log,
                  const float* __restrict__ Dd, const float* __restrict__ xz,
                  const float* __restrict__ hin, _Float16* __restrict__ yh) {
    __shared__ __align__(16) float sbc[CHUNK * 32];
    int i = blockIdx.x * 256 + threadIdx.x;        // NCH*DINNER
    int e = i & (DINNER - 1), c = i >> 10;
    stage_bc_tile(dbc, c, sbc, threadIdx.x);

    float An[DSTATE], h[DSTATE];
    size_t o = ((size_t)c * DINNER + e) * DSTATE;
#pragma unroll
    for (int n = 0; n < DSTATE; ++n) {
        An[n] = -expf(A_log[(size_t)e * DSTATE + n]);
        h[n] = hin[o + n];
    }
    const float De = Dd[e];
    const int tbase = c * CHUNK;
    for (int tt = 0; tt < CHUNK; ++tt) {
        const int t = tbase + tt;
        const float d  = delta[(size_t)t * DINNER + e];
        const float uu = u[(size_t)t * DINNER + e];
        const float du = d * uu;
        const float* Bv = sbc + tt * 32;           // B in LDS
        const float* Cv = Bv + DSTATE;             // C in LDS
        float y = 0.f;
#pragma unroll
        for (int n = 0; n < DSTATE; ++n) {
            float dA = expf(d * An[n]);
            h[n] = dA * h[n] + du * Bv[n];
            y += h[n] * Cv[n];
        }
        y += De * uu;
        const float z = xz[(size_t)t * (2 * DINNER) + DINNER + e];
        y *= dev_silu(z);
        yh[(size_t)t * DINNER + e] = (_Float16)y;
    }
}

// ---------------------------------------------------------------------------
// attention pooling tail
// ---------------------------------------------------------------------------
__global__ void k_scores(const float* __restrict__ t1, const float* __restrict__ w2,
                         const float* __restrict__ b2, float* __restrict__ scores) {
    int t = blockIdx.x * 256 + threadIdx.x;
    if (t >= NTOK) return;
    float s = b2[0];
    for (int k = 0; k < ATTND; ++k) s += t1[(size_t)t * ATTND + k] * w2[k];
    scores[t] = s;
}

__global__ __launch_bounds__(256)
void k_softmax(const float* __restrict__ s, float* __restrict__ w) {
    __shared__ float red[256];
    const int tid = threadIdx.x;
    float m = -1e30f;
    for (int i = tid; i < NTOK; i += 256) m = fmaxf(m, s[i]);
    red[tid] = m; __syncthreads();
    for (int off = 128; off > 0; off >>= 1) { if (tid < off) red[tid] = fmaxf(red[tid], red[tid + off]); __syncthreads(); }
    float mx = red[0]; __syncthreads();
    float sum = 0.f;
    for (int i = tid; i < NTOK; i += 256) sum += expf(s[i] - mx);
    red[tid] = sum; __syncthreads();
    for (int off = 128; off > 0; off >>= 1) { if (tid < off) red[tid] += red[tid + off]; __syncthreads(); }
    float inv = 1.f / red[0];
    for (int i = tid; i < NTOK; i += 256) w[i] = expf(s[i] - mx) * inv;
}

__global__ void k_pool(const float* __restrict__ w, const float* __restrict__ hl,
                       float* __restrict__ pooled) {
    int d = blockIdx.x * 256 + threadIdx.x;        // DMODEL
    float acc = 0.f;
    for (int t = 0; t < NTOK; ++t) acc += w[t] * hl[(size_t)t * DMODEL + d];
    pooled[d] = acc;
}

__global__ void k_cls(const float* __restrict__ pooled, const float* __restrict__ cls_w,
                      const float* __restrict__ cls_b, float* __restrict__ out) {
    int c = threadIdx.x;
    if (c < 2) {
        float a = cls_b[c];
        for (int d = 0; d < DMODEL; ++d) a += pooled[d] * cls_w[d * 2 + c];
        out[c] = a;
    }
}

// ---------------------------------------------------------------------------
// launcher
// ---------------------------------------------------------------------------
extern "C" void kernel_launch(void* const* d_in, const int* in_sizes, int n_in,
                              void* d_out, int out_size, void* d_ws, size_t ws_size,
                              hipStream_t stream) {
    (void)in_sizes; (void)n_in; (void)out_size; (void)ws_size;
    const float* x       = (const float*)d_in[0];
    const float* pos     = (const float*)d_in[1];
    const float* fc1_w   = (const float*)d_in[2];
    const float* fc1_b   = (const float*)d_in[3];
    const float* pos_w   = (const float*)d_in[4];
    const float* pos_b   = (const float*)d_in[5];
    const float* in_proj = (const float*)d_in[6];
    const float* conv_w  = (const float*)d_in[7];
    const float* conv_b  = (const float*)d_in[8];
    const float* x_proj  = (const float*)d_in[9];
    const float* dt_w    = (const float*)d_in[10];
    const float* dt_b    = (const float*)d_in[11];
    const float* A_log   = (const float*)d_in[12];
    const float* Dd      = (const float*)d_in[13];
    const float* out_w   = (const float*)d_in[14];
    const float* rms_w   = (const float*)d_in[15];
    const float* norm_w  = (const float*)d_in[16];
    const float* norm_b  = (const float*)d_in[17];
    const float* attn_w1 = (const float*)d_in[18];
    const float* attn_b1 = (const float*)d_in[19];
    const float* attn_w2 = (const float*)d_in[20];
    const float* attn_b2 = (const float*)d_in[21];
    const float* cls_w   = (const float*)d_in[22];
    const float* cls_b   = (const float*)d_in[23];
    float* out = (float*)d_out;

    // workspace arena (256B-aligned slices)
    char* p = (char*)d_ws;
    auto alloc = [&](size_t bytes) { char* r = p; p += (bytes + 255) & ~(size_t)255; return r; };
    _Float16* xh    = (_Float16*)alloc((size_t)NTOK * INDIM * 2);
    _Float16* fc1T  = (_Float16*)alloc((size_t)DMODEL * INDIM * 2);
    _Float16* inpT  = (_Float16*)alloc((size_t)NLAYER * 2 * DINNER * DMODEL * 2);
    _Float16* xpT   = (_Float16*)alloc((size_t)NLAYER * 64 * DINNER * 2);
    _Float16* dtT   = (_Float16*)alloc((size_t)NLAYER * DINNER * DTRANK * 2);
    _Float16* owT   = (_Float16*)alloc((size_t)NLAYER * DMODEL * DINNER * 2);
    _Float16* a1T   = (_Float16*)alloc((size_t)ATTND * DMODEL * 2);
    float*    h     = (float*)   alloc((size_t)NTOK * DMODEL * 4);
    _Float16* hn    = (_Float16*)alloc((size_t)NTOK * DMODEL * 2);
    float*    xz    = (float*)   alloc((size_t)NTOK * 2 * DINNER * 4);
    float*    u     = (float*)   alloc((size_t)NTOK * DINNER * 4);
    _Float16* uh    = (_Float16*)alloc((size_t)NTOK * DINNER * 2);
    float*    dbc   = (float*)   alloc((size_t)NTOK * 64 * 4);
    _Float16* dtrh  = (_Float16*)alloc((size_t)NTOK * DTRANK * 2);
    float*    delta = (float*)   alloc((size_t)NTOK * DINNER * 4);
    float*    chA   = (float*)   alloc((size_t)NCH * DINNER * DSTATE * 4);
    float*    chB   = (float*)   alloc((size_t)NCH * DINNER * DSTATE * 4);
    float*    hin   = (float*)   alloc((size_t)NCH * DINNER * DSTATE * 4);
    _Float16* yh    = (_Float16*)alloc((size_t)NTOK * DINNER * 2);
    float*    hl    = (float*)   alloc((size_t)NTOK * DMODEL * 4);
    _Float16* hlh   = (_Float16*)alloc((size_t)NTOK * DMODEL * 2);
    float*    t1    = (float*)   alloc((size_t)NTOK * ATTND * 4);
    float*    scr   = (float*)   alloc((size_t)NTOK * 4);
    float*    aw    = (float*)   alloc((size_t)NTOK * 4);
    float*    pool  = (float*)   alloc((size_t)DMODEL * 4);

    auto blocks = [](size_t n) { return (unsigned)((n + 255) / 256); };
    auto ggrid  = [](int N, int M) { return dim3((unsigned)((N + 127) / 128), (unsigned)(M / 32)); };

    // --- per-call weight conversion/transposition to f16 ---
    k_cvt<<<blocks((size_t)NTOK * INDIM), 256, 0, stream>>>(x, xh, NTOK * INDIM);
    k_cvt_transpose<<<blocks((size_t)INDIM * DMODEL), 256, 0, stream>>>(fc1_w, fc1T, INDIM, DMODEL);
    for (int l = 0; l < NLAYER; ++l) {
        k_cvt_transpose<<<blocks((size_t)DMODEL * 2 * DINNER), 256, 0, stream>>>(
            in_proj + (size_t)l * DMODEL * 2 * DINNER, inpT + (size_t)l * 2 * DINNER * DMODEL,
            DMODEL, 2 * DINNER);
        k_cvt_transpose<<<blocks((size_t)DINNER * 64), 256, 0, stream>>>(
            x_proj + (size_t)l * DINNER * 64, xpT + (size_t)l * 64 * DINNER, DINNER, 64);
        k_cvt_transpose<<<blocks((size_t)DTRANK * DINNER), 256, 0, stream>>>(
            dt_w + (size_t)l * DTRANK * DINNER, dtT + (size_t)l * DINNER * DTRANK, DTRANK, DINNER);
        k_cvt_transpose<<<blocks((size_t)DINNER * DMODEL), 256, 0, stream>>>(
            out_w + (size_t)l * DINNER * DMODEL, owT + (size_t)l * DMODEL * DINNER, DINNER, DMODEL);
    }
    k_cvt_transpose<<<blocks((size_t)DMODEL * ATTND), 256, 0, stream>>>(attn_w1, a1T, DMODEL, ATTND);

    // --- stem: h = gelu(x @ fc1 + b) + pos embedding ---
    k_gemm_f16<<<ggrid(DMODEL, NTOK), 128, 0, stream>>>(xh, fc1T, h, fc1_b, NTOK, DMODEL, INDIM, 0);
    k_gelu_posadd<<<blocks((size_t)NTOK * DMODEL), 256, 0, stream>>>(h, pos, pos_w, pos_b);

    // --- Mamba layers ---
    for (int l = 0; l < NLAYER; ++l) {
        k_rmsnorm_f16<<<NTOK, 256, 0, stream>>>(h, rms_w + (size_t)l * DMODEL, hn);
        k_gemm_f16<<<ggrid(2 * DINNER, NTOK), 128, 0, stream>>>(
            hn, inpT + (size_t)l * 2 * DINNER * DMODEL, xz, nullptr, NTOK, 2 * DINNER, DMODEL, 0);
        k_conv_silu<<<blocks((size_t)NTOK * DINNER), 256, 0, stream>>>(
            xz, conv_w + (size_t)l * DINNER * DCONV, conv_b + (size_t)l * DINNER, u, uh);
        k_gemm_f16<<<ggrid(64, NTOK), 128, 0, stream>>>(
            uh, xpT + (size_t)l * 64 * DINNER, dbc, nullptr, NTOK, 64, DINNER, 0);
        k_extract_dtr<<<blocks((size_t)NTOK * DTRANK), 256, 0, stream>>>(dbc, dtrh);
        k_gemm_f16<<<ggrid(DINNER, NTOK), 128, 0, stream>>>(
            dtrh, dtT + (size_t)l * DINNER * DTRANK, delta, dt_b + (size_t)l * DINNER,
            NTOK, DINNER, DTRANK, 0);
        k_softplus<<<blocks((size_t)NTOK * DINNER), 256, 0, stream>>>(delta, NTOK * DINNER);

        const float* Al = A_log + (size_t)l * DINNER * DSTATE;
        k_scan_chunks<<<blocks((size_t)NCH * DINNER), 256, 0, stream>>>(delta, u, dbc, Al, chA, chB);
        k_scan_prefix<<<blocks(DINNER), 256, 0, stream>>>(chA, chB, hin);
        k_scan_apply<<<blocks((size_t)NCH * DINNER), 256, 0, stream>>>(
            delta, u, dbc, Al, Dd + (size_t)l * DINNER, xz, hin, yh);

        // residual: h += y @ out_w
        k_gemm_f16<<<ggrid(DMODEL, NTOK), 128, 0, stream>>>(
            yh, owT + (size_t)l * DMODEL * DINNER, h, nullptr, NTOK, DMODEL, DINNER, 1);
    }

    // --- tail: LayerNorm, gated-attention MIL pooling, classifier ---
    k_layernorm<<<NTOK, 256, 0, stream>>>(h, norm_w, norm_b, hl, hlh);
    k_gemm_f16<<<ggrid(ATTND, NTOK), 128, 0, stream>>>(hlh, a1T, t1, attn_b1, NTOK, ATTND, DMODEL, 0);
    k_tanh_ew<<<blocks((size_t)NTOK * ATTND), 256, 0, stream>>>(t1, NTOK * ATTND);
    k_scores<<<blocks(NTOK), 256, 0, stream>>>(t1, attn_w2, attn_b2, scr);
    k_softmax<<<1, 256, 0, stream>>>(scr, aw);
    k_pool<<<DMODEL / 256, 256, 0, stream>>>(aw, hl, pool);
    k_cls<<<1, 32, 0, stream>>>(pool, cls_w, cls_b, out);
}
